// KGramEmbeddingMLP_91018946937244
// MI455X (gfx1250) — matrix operand order repaired and checked by reading, not compiled, not run on hardware
//
#include <hip/hip_runtime.h>

// ---------------------------------------------------------------------------
// KGram Embedding MLP, MI455X (gfx1250, wave32, WMMA)
//
// Roofline: mandatory HBM traffic = 412MB (one-hot scan) + 206MB (W3)
//           + 51MB (out) ~= 670MB -> ~29us @ 23.3TB/s.  The 27 GFLOP of GEMM
//           runs on v_wmma_f32_16x16x32_bf16 so compute is negligible.
// ---------------------------------------------------------------------------

#define VOCAB  50257
#define KCTX   8
#define EMBED  64
#define HIDDEN 1024
#define BATCH  256

typedef __attribute__((ext_vector_type(16))) __bf16 v16bf;
typedef __attribute__((ext_vector_type(8)))  __bf16 v8bf;
typedef __attribute__((ext_vector_type(8)))  float  v8f;

// ---------------------------------------------------------------------------
// Kernel 1: per (b,k) slot, find the one-hot index (coalesced scan of 50257
// floats) and gather the 64-wide embedding row into x0 as bf16.
// Grid: BATCH*KCTX blocks of 256 threads.
// ---------------------------------------------------------------------------
__global__ void onehot_gather(const float* __restrict__ ctx,
                              const float* __restrict__ embed_w,
                              __bf16* __restrict__ x0)
{
    __shared__ int s_idx;
    const int bk  = blockIdx.x;            // b*KCTX + k
    const int tid = threadIdx.x;
    if (tid == 0) s_idx = 0;
    __syncthreads();

    const float* row = ctx + (size_t)bk * VOCAB;
    for (int v = tid; v < VOCAB; v += 256) {
        if (row[v] != 0.0f) s_idx = v;     // exactly one nonzero per row
    }
    __syncthreads();

    const int idx = s_idx;
    if (tid < EMBED) {
        x0[(size_t)bk * EMBED + tid] =
            (__bf16)embed_w[(size_t)idx * EMBED + tid];
    }
}

// pack two f32 -> one dword holding 2 bf16 (RNE), single VALU op
__device__ __forceinline__ unsigned int pk_bf16(float lo, float hi)
{
    unsigned int r;
    asm("v_cvt_pk_bf16_f32 %0, %1, %2" : "=v"(r) : "v"(lo), "v"(hi));
    return r;
}

// ---------------------------------------------------------------------------
// Kernel 2: GEMM  out[M,N] = act(A[M,K] @ W[K,N] + bias)
//   A : bf16 row-major (activations), M == 256
//   W : fp32 row-major (weights), converted to bf16 in-register
//   Per wave: one 16-wide N column (nt); MLOOP M-tiles in fp32 accumulators;
//   B tile loaded once per 32-deep K step, reused across all M tiles ->
//   W3 streamed from HBM exactly once (A stays hot in the 192MB L2).
//
// OOB handling: WMMA column n depends only on B column n (lane-local), so
// tail lanes load a clamped address (no EXEC-mask churn, no zeroing) and we
// simply skip their stores. EXEC stays all-ones through every WMMA.
//
// VGPR layouts per CDNA5 ISA 7.12.2:
//   A (16-bit 16x32): half 0 -> K 0..7 / 16..23, half 1 -> K 8..15 / 24..31
//   B (16-bit 32x16): half 0 -> K 0..7 / 16..23, half 1 -> K 8..15 / 24..31
//   C/D f32 16x16  :  VGPR r -> m = half*8 + r,   n = lane & 15
// ---------------------------------------------------------------------------
template <int MLOOP, bool SILU, bool OUT_BF16>
__global__ void gemm_wmma(const __bf16* __restrict__ A,   // [256, Kd] bf16
                          const float*  __restrict__ W,   // [Kd, N]  fp32
                          const float*  __restrict__ bias,// [N]
                          void* __restrict__ outp,        // [256, N]
                          int Kd, int N)
{
    const int lane = threadIdx.x & 31;
    const int wid  = threadIdx.x >> 5;          // 8 waves per block
    const int nt   = blockIdx.x * 8 + wid;      // N-tile (16 columns)
    const int n    = nt * 16 + (lane & 15);
    const int half = lane >> 4;
    const int mt0  = blockIdx.y * MLOOP;        // first M-tile for this wave
    const int nc   = (n < N) ? n : (N - 1);     // clamped: keeps loads legal

    v8f acc[MLOOP];
#pragma unroll
    for (int j = 0; j < MLOOP; ++j)
#pragma unroll
        for (int r = 0; r < 8; ++r) acc[j][r] = 0.0f;

    // running W pointer for this lane-half's 16 K-rows of column nc
    const float* wptr = W + (size_t)(half * 8) * N + nc;
    const size_t wstep = (size_t)32 * N;
    // A base for this lane's M row within tile 0 of this wave's sweep
    const __bf16* abase =
        A + (size_t)(mt0 * 16 + (lane & 15)) * Kd + half * 8;

    for (int kb = 0; kb < Kd; kb += 32) {
        // ---- B tile: 16 coalesced dword loads, packed pairwise to bf16
        float f[16];
#pragma unroll
        for (int i = 0; i < 16; ++i)
            f[i] = wptr[(size_t)(i < 8 ? i : i + 8) * N];
        wptr += wstep;
        if (kb + 32 < Kd)
            __builtin_prefetch(wptr, 0, 1);

        union { v16bf v; unsigned int u[8]; } bu;
#pragma unroll
        for (int i = 0; i < 8; ++i)
            bu.u[i] = pk_bf16(f[2 * i], f[2 * i + 1]);
        const v16bf b = bu.v;

        // ---- sweep M tiles, B stays resident in registers
#pragma unroll
        for (int j = 0; j < MLOOP; ++j) {
            const __bf16* ap = abase + (size_t)j * 16 * Kd + kb;
            const v8bf alo = *(const v8bf*)(ap);       // K = kb+half*8 +0..7
            const v8bf ahi = *(const v8bf*)(ap + 16);  // K = kb+16+half*8+0..7
            v16bf a;
#pragma unroll
            for (int i = 0; i < 8; ++i) { a[i] = alo[i]; a[8 + i] = ahi[i]; }

            acc[j] = __builtin_amdgcn_wmma_f32_16x16x32_bf16(
                false, a, false, b, (short)0, acc[j], false, false);
        }
    }

    if (n >= N) return;                      // tail columns: nothing to store
    const float bn = bias[n];
#pragma unroll
    for (int j = 0; j < MLOOP; ++j) {
#pragma unroll
        for (int r = 0; r < 8; ++r) {
            const int m = (mt0 + j) * 16 + half * 8 + r;
            float v = acc[j][r] + bn;
            if (SILU) v = v / (1.0f + __expf(-v));
            if (OUT_BF16) ((__bf16*)outp)[(size_t)m * N + n] = (__bf16)v;
            else          ((float*) outp)[(size_t)m * N + n] = v;
        }
    }
}

// ---------------------------------------------------------------------------
// Launch
// ---------------------------------------------------------------------------
extern "C" void kernel_launch(void* const* d_in, const int* in_sizes, int n_in,
                              void* d_out, int out_size, void* d_ws, size_t ws_size,
                              hipStream_t stream)
{
    const float* ctx = (const float*)d_in[0];   // [256, 8*50257]
    const float* emb = (const float*)d_in[1];   // [50257, 64]
    const float* W1  = (const float*)d_in[2];   // [512, 1024]
    const float* b1  = (const float*)d_in[3];
    const float* W2  = (const float*)d_in[4];   // [1024, 1024]
    const float* b2  = (const float*)d_in[5];
    const float* W3  = (const float*)d_in[6];   // [1024, 50257]
    const float* b3  = (const float*)d_in[7];
    float* out = (float*)d_out;                 // [256, 50257]

    // workspace: bf16 activations (all 16B-aligned offsets)
    char* ws = (char*)d_ws;
    __bf16* x0 = (__bf16*)(ws);                          // 256* 512*2 = 256KB
    __bf16* x1 = (__bf16*)(ws + 256 * 1024);             // 256*1024*2 = 512KB
    __bf16* x2 = (__bf16*)(ws + 256 * 1024 + 512 * 1024);// 512KB

    // 1) one-hot -> index -> embedding gather (bf16)
    onehot_gather<<<BATCH * KCTX, 256, 0, stream>>>(ctx, emb, x0);

    // 2) layer 1: [256,512] @ [512,1024] + b1, SiLU  -> x1 (bf16)
    //    8 waves/block * 16 cols = 128 cols/block; MLOOP=4 -> 4 y-blocks
    dim3 g12(HIDDEN / 128, (BATCH / 16) / 4);
    gemm_wmma<4, true, true><<<g12, 256, 0, stream>>>(x0, W1, b1, x1,
                                                      KCTX * EMBED, HIDDEN);

    // 3) layer 2: [256,1024] @ [1024,1024] + b2, SiLU -> x2 (bf16)
    gemm_wmma<4, true, true><<<g12, 256, 0, stream>>>(x1, W2, b2, x2,
                                                      HIDDEN, HIDDEN);

    // 4) layer 3: [256,1024] @ [1024,50257] + b3 -> out (fp32)
    //    MLOOP=16 covers all of M per block -> W3 read from HBM exactly once
    dim3 g3((VOCAB + 127) / 128, 1);
    gemm_wmma<16, false, false><<<g3, 256, 0, stream>>>(x2, W3, b3, out,
                                                        HIDDEN, VOCAB);
}
